// HMMCell_65309272703482
// MI455X (gfx1250) — matrix operand order repaired and checked by reading, compiled, not run on hardware
//
#include <hip/hip_runtime.h>
#include <hip/hip_bf16.h>
#include <cstdint>
#include <cstddef>

// Problem constants (from reference)
#define HN   2048     // hidden states
#define HS   64       // emission alphabet
#define HT   16384    // timesteps
#define NP1  2049     // N+1 (output row width)
#define RWG  128      // recurrence workgroups (each owns 16 rows of A^T in LDS)
#define RROWS 16      // rows of A^T per workgroup
#define APITCH 17     // padded LDS row pitch (TDM pad: 16 DWORDs + 1 DWORD)

typedef _Float16 half_t;
typedef __attribute__((ext_vector_type(16))) _Float16 v16h;
typedef __attribute__((ext_vector_type(8)))  _Float16 v8h;
typedef __attribute__((ext_vector_type(8)))  float    v8f;

// ---------------------------------------------------------------------------
// Device-wide barrier: monotonically increasing counter, one arrival per WG.
// ---------------------------------------------------------------------------
__device__ __forceinline__ void grid_sync(unsigned* cnt, unsigned target) {
  __threadfence();            // make this wave's global stores visible
  __syncthreads();
  if (threadIdx.x == 0) {
    __hip_atomic_fetch_add(cnt, 1u, __ATOMIC_ACQ_REL, __HIP_MEMORY_SCOPE_AGENT);
    while (__hip_atomic_load(cnt, __ATOMIC_ACQUIRE, __HIP_MEMORY_SCOPE_AGENT) < target) {
      __builtin_amdgcn_s_sleep(2);
    }
  }
  __syncthreads();
}

// ---------------------------------------------------------------------------
// Zero the barrier counter (workspace is poisoned by harness each run)
// ---------------------------------------------------------------------------
__global__ void zero_counter_kernel(unsigned* cnt) { *cnt = 0u; }

// ---------------------------------------------------------------------------
// Row-wise softmax, generic width (transition [N x N] rows, init [1 x N])
// ---------------------------------------------------------------------------
__global__ void __launch_bounds__(256)
row_softmax_kernel(const float* __restrict__ in, float* __restrict__ out, int cols) {
  __shared__ float red[256];
  const int row = blockIdx.x;
  const float* r = in + (size_t)row * cols;
  float m = -__builtin_inff();
  for (int j = threadIdx.x; j < cols; j += 256) m = fmaxf(m, r[j]);
  red[threadIdx.x] = m;
  __syncthreads();
  for (int s = 128; s > 0; s >>= 1) {
    if (threadIdx.x < s) red[threadIdx.x] = fmaxf(red[threadIdx.x], red[threadIdx.x + s]);
    __syncthreads();
  }
  m = red[0];
  __syncthreads();
  float acc = 0.f;
  for (int j = threadIdx.x; j < cols; j += 256) acc += __expf(r[j] - m);
  red[threadIdx.x] = acc;
  __syncthreads();
  for (int s = 128; s > 0; s >>= 1) {
    if (threadIdx.x < s) red[threadIdx.x] += red[threadIdx.x + s];
    __syncthreads();
  }
  const float inv = 1.f / red[0];
  float* o = out + (size_t)row * cols;
  for (int j = threadIdx.x; j < cols; j += 256) o[j] = __expf(r[j] - m) * inv;
}

// ---------------------------------------------------------------------------
// Emission softmax (rows of 64) -> f16, stored TRANSPOSED as BhT[s][n] so the
// WMMA B-fragment loads are coalesced across lanes (consecutive n). One wave per row n.
// ---------------------------------------------------------------------------
__global__ void __launch_bounds__(256)
emis_softmax_kernel(const float* __restrict__ ek, half_t* __restrict__ BhT) {
  const int n    = blockIdx.x * 8 + (threadIdx.x >> 5);
  const int lane = threadIdx.x & 31;
  const float x0 = ek[(size_t)n * HS + lane];
  const float x1 = ek[(size_t)n * HS + 32 + lane];
  float m = fmaxf(x0, x1);
  for (int o = 16; o > 0; o >>= 1) m = fmaxf(m, __shfl_xor(m, o, 32));
  const float e0 = __expf(x0 - m);
  const float e1 = __expf(x1 - m);
  float s = e0 + e1;
  for (int o = 16; o > 0; o >>= 1) s += __shfl_xor(s, o, 32);
  const float inv = 1.f / s;
  BhT[(size_t)lane        * HN + n] = (half_t)(e0 * inv);
  BhT[(size_t)(lane + 32) * HN + n] = (half_t)(e1 * inv);
}

// ---------------------------------------------------------------------------
// f32 -> f16 conversion of the observation stream
// ---------------------------------------------------------------------------
__global__ void __launch_bounds__(256)
cvt_f16_kernel(const float* __restrict__ in, half_t* __restrict__ out) {
  const size_t i = (size_t)blockIdx.x * 256 + threadIdx.x;
  out[i] = (half_t)in[i];
}

// ---------------------------------------------------------------------------
// E[t][n] = sum_s Xh[t][s] * B[n][s]  via V_WMMA_F32_16X16X32_F16.
// (1024 x 128) 16x16 tiles, 8 waves (one tile each) per block, K = 2x32.
// ---------------------------------------------------------------------------
__global__ void __launch_bounds__(256)
egemm_wmma_kernel(const half_t* __restrict__ Xh, const half_t* __restrict__ BhT,
                  float* __restrict__ E) {
  const int wave = threadIdx.x >> 5;
  const int lane = threadIdx.x & 31;
  const int tile = blockIdx.x * 8 + wave;
  const int t0   = (tile >> 7) << 4;                   // tile / (HN/16) * 16
  const int n0   = (tile & 127) << 4;                  // tile % (HN/16) * 16
  const int lrow = lane & 15;
  const int kg   = (lane >> 4) << 3;                   // 0 or 8
  const int klane = (lane & 15) | ((lane >> 4) << 4);  // 0..31

  v8f c = {};
#pragma unroll
  for (int kb = 0; kb < HS; kb += 32) {
    const half_t* xr = Xh + (size_t)(t0 + lrow) * HS + kb;
    const v8h alo = *(const v8h*)(xr + kg);
    const v8h ahi = *(const v8h*)(xr + 16 + kg);
    v16h a;
#pragma unroll
    for (int h = 0; h < 8; ++h) { a[h] = alo[h]; a[h + 8] = ahi[h]; }
    v16h b;
#pragma unroll
    for (int h = 0; h < 8; ++h) {
      b[h]     = BhT[(size_t)(kb + kg + h) * HN + n0 + lrow];
      b[h + 8] = BhT[(size_t)(kb + 16 + kg + h) * HN + n0 + lrow];
    }
    // (neg_a, A, neg_b, B, c_mod, C, reuse_a, reuse_b)
    c = __builtin_amdgcn_wmma_f32_16x16x32_f16(false, a, false, b, (short)0, c,
                                               false, false);
  }
  const int n  = lane & 15;
  const int mb = (lane >> 4) << 3;
#pragma unroll
  for (int r = 0; r < 8; ++r)
    E[(size_t)(t0 + mb + r) * HN + (n0 + n)] = c[r];
}

// ---------------------------------------------------------------------------
// out[t*2049 + 0] = 0 for all t
// ---------------------------------------------------------------------------
__global__ void __launch_bounds__(256)
zero_col0_kernel(float* __restrict__ out) {
  const int t = blockIdx.x * 256 + threadIdx.x;
  out[(size_t)t * NP1] = 0.f;
}

// ---------------------------------------------------------------------------
// Persistent recurrence kernel. 128 WGs; WG b owns rows [16b, 16b+16) of A^T,
// fetched once by the Tensor Data Mover as a 2D tile (tile 16 x 2048, tensor
// stride 2048, 4B elements) with TDM padding -> 17-DWORD LDS pitch, which
// makes the inner-loop LDS pattern (bank = (17p + r) mod 64) conflict-free.
// Per step: async global->LDS stage of v_{t-1} from a 16B-aligned mirror,
// 16 dot products split 16 ways, LDS reduce, scale by E, one grid barrier.
// ---------------------------------------------------------------------------
__global__ void __launch_bounds__(256, 1)
recurrence_kernel(const float* __restrict__ A, const float* __restrict__ I,
                  const float* __restrict__ E, float* __restrict__ out,
                  float* __restrict__ vglob, unsigned* __restrict__ counter) {
  extern __shared__ float lds[];
  float* aslc = lds;                        // HN * APITCH floats (padded A^T slice)
  float* vbuf = aslc + (size_t)HN * APITCH; // HN floats (v_{t-1} stage, 16B aligned)
  float* part = vbuf + HN;                  // 256 floats (partial sums)

  const int tid = threadIdx.x;
  const int i0  = blockIdx.x * RROWS;

  // ---- One-time TDM load: A^T slice (columns i0..i0+15 of A) into LDS ----
  // D# group0: count=1 | lds_addr | global_addr | type=2
  // D# group1: data_size=4B, pad_enable, pad_interval=16dw, pad_amount=1dw,
  //            tensor_dim0=2048, tensor_dim1=2048, tile_dim0=16,
  //            tile_dim1=2048, tensor_dim0_stride=2048
  if (tid == 0) {
    const unsigned lds0 = (unsigned)(size_t)aslc;
    const unsigned long long ga = (unsigned long long)(size_t)(A + i0);
    const unsigned ga_lo = (unsigned)ga;
    const unsigned ga_hi = (unsigned)((ga >> 32) & 0x01FFFFFFull) | (2u << 30);
    asm volatile(
        "s_mov_b32 s60, 1\n\t"
        "s_mov_b32 s61, %0\n\t"
        "s_mov_b32 s62, %1\n\t"
        "s_mov_b32 s63, %2\n\t"
        "s_mov_b32 s64, 0x00D20000\n\t"  // data_size=4B | pad_en | interval=16dw
        "s_mov_b32 s65, 0x08000000\n\t"  // tensor_dim0 = 2048 (low16 -> [31:16])
        "s_mov_b32 s66, 0x08000000\n\t"  // tensor_dim0 hi=0, tensor_dim1 lo=2048
        "s_mov_b32 s67, 0x00100000\n\t"  // tensor_dim1 hi=0, tile_dim0=16
        "s_mov_b32 s68, 0x00000800\n\t"  // tile_dim1=2048, tile_dim2=0
        "s_mov_b32 s69, 0x00000800\n\t"  // tensor_dim0_stride = 2048
        "s_mov_b32 s70, 0\n\t"
        "s_mov_b32 s71, 0\n\t"
        "tensor_load_to_lds s[60:63], s[64:71]\n\t"
        "s_wait_tensorcnt 0x0"
        :: "s"(lds0), "s"(ga_lo), "s"(ga_hi)
        : "s60","s61","s62","s63","s64","s65","s66","s67","s68","s69","s70",
          "s71","memory");
  }

  // t = 0 : v_0 = E_0 * I   (alpha0 = e_0 -> R = I); mirror into vglob[0]
  if (tid < RROWS) {
    const int i = i0 + tid;
    const float v0 = E[i] * I[i];
    out[1 + i]  = v0;
    vglob[i]    = v0;
  }
  grid_sync(counter, RWG);

  const int r = tid >> 4;               // output row within slice (0..15)
  const int p = tid & 15;               // partition of the K loop (0..15)
  for (int t = 1; t < HT; ++t) {
    // Async-stage v_{t-1} (8 KB, 16B-aligned mirror) into LDS: 2x b128/lane.
    {
      const float* gsrc = vglob + (size_t)((t - 1) & 1) * HN + tid * 8;
      const unsigned l0 = (unsigned)(size_t)(vbuf + tid * 8);
      asm volatile(
          "global_load_async_to_lds_b128 %0, %1, off\n\t"
          "global_load_async_to_lds_b128 %0, %1, off offset:16\n\t"
          "s_wait_asynccnt 0x0"
          :: "v"(l0), "v"(gsrc) : "memory");
    }
    __syncthreads();

    float acc = 0.f;
#pragma unroll 8
    for (int jj = 0; jj < HN / 16; ++jj) {
      const int j = p + (jj << 4);
      acc = __builtin_fmaf(aslc[j * APITCH + r], vbuf[j], acc);
    }
    part[tid] = acc;
    __syncthreads();

    if (tid < RROWS) {
      float s = 0.f;
#pragma unroll
      for (int q = 0; q < 16; ++q) s += part[tid * 16 + q];
      const int i = i0 + tid;
      const float vt = E[(size_t)t * HN + i] * s;
      out[(size_t)t * NP1 + 1 + i]        = vt;
      vglob[(size_t)(t & 1) * HN + i]     = vt;
    }
    grid_sync(counter, (unsigned)(t + 1) * RWG);
  }
}

// ---------------------------------------------------------------------------
// Host orchestration
// ---------------------------------------------------------------------------
extern "C" void kernel_launch(void* const* d_in, const int* in_sizes, int n_in,
                              void* d_out, int out_size, void* d_ws, size_t ws_size,
                              hipStream_t stream) {
  (void)in_sizes; (void)n_in; (void)out_size; (void)ws_size;
  const float* x  = (const float*)d_in[0];   // [1, T, S]
  const float* ek = (const float*)d_in[1];   // [N, S]
  const float* tk = (const float*)d_in[2];   // [N, N]
  const float* ik = (const float*)d_in[3];   // [N]
  float* out = (float*)d_out;                // [1, T, N+1]

  // Workspace carve-up (all offsets 256B-aligned)
  char*     ws      = (char*)d_ws;
  unsigned* counter = (unsigned*)ws;                       // 4 B (+pad)
  float*    A       = (float*)(ws + 256);                  // N*N f32 = 16 MB
  float*    I       = A + (size_t)HN * HN;                 // N f32
  half_t*   BhT     = (half_t*)(I + HN);                   // S*N f16 (transposed)
  half_t*   Xh      = BhT + (size_t)HS * HN;               // T*S f16
  float*    E       = (float*)(Xh + (size_t)HT * HS);      // T*N f32 = 128 MB
  float*    vglob   = E + (size_t)HT * HN;                 // 2*N f32 (v mirror)

  zero_counter_kernel<<<1, 1, 0, stream>>>(counter);
  row_softmax_kernel<<<HN, 256, 0, stream>>>(tk, A, HN);   // A = softmax rows
  row_softmax_kernel<<<1, 256, 0, stream>>>(ik, I, HN);    // I = softmax init
  emis_softmax_kernel<<<HN / 8, 256, 0, stream>>>(ek, BhT);
  cvt_f16_kernel<<<(HT * HS) / 256, 256, 0, stream>>>(x, Xh);
  egemm_wmma_kernel<<<(HT / 16) * (HN / 16) / 8, 256, 0, stream>>>(Xh, BhT, E);
  zero_col0_kernel<<<HT / 256, 256, 0, stream>>>(out);

  const size_t ldsBytes = (size_t)HN * APITCH * sizeof(float)  // padded A^T slice
                        + (size_t)HN * sizeof(float)           // v stage
                        + 256 * sizeof(float);                 // partials
  recurrence_kernel<<<RWG, 256, ldsBytes, stream>>>(A, I, E, out, vglob, counter);
}
